// Llama3_Attention_59674275610739
// MI455X (gfx1250) — matrix operand order repaired and checked by reading, not compiled, not run on hardware
//
#include <hip/hip_runtime.h>

// ---------------------------------------------------------------------------
// Llama3 GQA attention for MI455X (gfx1250, wave32, WMMA 16x16x32 f16).
//   B=2, T=2048, E=4096, H=32, head_dim=128, KV heads=8.
// Pipeline: fp32->f16 convert -> Q/K/V projection GEMMs (f16 WMMA, f32 acc,
//           async-to-LDS double-buffered tiles) -> fused flash-attention
//           (WMMA QK^T and PV, online softmax, async K tiles) -> output
//           projection GEMM (fp32 out).
// ---------------------------------------------------------------------------

#define BQ   2
#define TQ   2048
#define EQ   4096
#define HQ   32
#define DQ   128
#define HKQ  8
#define EKVQ 1024
#define MTQ  (BQ * TQ)   // 4096 rows of the "token" GEMMs

typedef __attribute__((ext_vector_type(16))) _Float16 v16h;
typedef __attribute__((ext_vector_type(8)))  float    v8f;

// ---------------------------------------------------------------------------
// WMMA wrapper: D = A(16x32 f16) * B(32x16 f16) + C(16x16 f32)
// ---------------------------------------------------------------------------
__device__ __forceinline__ v8f wmma_f16(v16h a, v16h b, v8f c) {
  return __builtin_amdgcn_wmma_f32_16x16x32_f16(
      /*neg_a=*/false, a, /*neg_b=*/false, b,
      /*c_mod=*/(short)0, c, /*reuse_a=*/false, /*reuse_b=*/false);
}

// ---------------------------------------------------------------------------
// CDNA5 async global->LDS copy (ISA 10.x / CDNA5 ISA, ASYNCcnt).
// Each active lane copies 16B from its global address to its LDS address.
// Generic LDS pointers carry the 32-bit LDS offset in their low DWORD.
// ---------------------------------------------------------------------------
__device__ __forceinline__ void async_copy_b128(const _Float16* gsrc,
                                                _Float16* ldst) {
  asm volatile("global_load_async_to_lds_b128 %0, %1, off"
               :: "v"((unsigned)(unsigned long long)ldst), "v"(gsrc)
               : "memory");
}
__device__ __forceinline__ void wait_async(void) {     // drain ASYNCcnt
  asm volatile("s_wait_asynccnt 0x0" ::: "memory");
}
__device__ __forceinline__ void wait_async_le4(void) { // keep 4 in flight
  asm volatile("s_wait_asynccnt 0x4" ::: "memory");
}

// ---------------------------------------------------------------------------
// Fragment loaders from LDS (wave32; contiguous k order used by both A and B, not the ISA table order).
// A (16x32, 16-bit): lanes 0-15 hold M=lane, K=0..15 contiguous;
//                    lanes 16-31 hold M=lane-16, K=16..31 contiguous.
// B (32x16, 16-bit): lanes 0-15 hold N=lane,   K=0..15 contiguous;
//                    lanes 16-31 hold N=lane-16, K=16..31 contiguous.
// ---------------------------------------------------------------------------
__device__ __forceinline__ v16h frag_a_lds(const _Float16* smem, int stride,
                                           int row0, int k0) {
  const int lane = threadIdx.x & 31;
  const int r    = lane & 15;
  const int kb   = (lane < 16) ? 0 : 16;
  union { unsigned int u[8]; v16h h; } f;
  const _Float16* base = smem + (size_t)(row0 + r) * stride + k0 + kb;
#pragma unroll
  for (int p = 0; p < 8; ++p) {
    const int k = 2 * p;
    f.u[p] = *(const unsigned int*)(base + k);   // ds_load_b32
  }
  return f.h;
}

__device__ __forceinline__ v16h frag_b_lds(const _Float16* smem, int stride,
                                           int col0, int k0) {
  const int lane = threadIdx.x & 31;
  const int c    = lane & 15;
  const int koff = (lane < 16) ? 0 : 16;
  const uint4* p = (const uint4*)(smem + (size_t)(col0 + c) * stride + k0 + koff);
  union { uint4 u[2]; v16h h; } f;
  f.u[0] = p[0];                                 // ds_load_b128 x2
  f.u[1] = p[1];
  return f.h;
}

// ---------------------------------------------------------------------------
// fp32 -> f16 conversion (our n are multiples of 1024)
// ---------------------------------------------------------------------------
__global__ __launch_bounds__(256) void cvt_f32_to_f16(
    const float* __restrict__ src, _Float16* __restrict__ dst, int n) {
  const int i = (blockIdx.x * 256 + threadIdx.x) * 4;
  if (i + 3 < n) {
    const float4 v = *(const float4*)(src + i);
    dst[i + 0] = (_Float16)v.x;
    dst[i + 1] = (_Float16)v.y;
    dst[i + 2] = (_Float16)v.z;
    dst[i + 3] = (_Float16)v.w;
  }
}

// ---------------------------------------------------------------------------
// C[M,N] = A[M,K] @ W[N,K]^T + bias[N]
//   f16 A/W, f32 accumulate; output f16 (staging) or f32 (final result).
// Block: 256 threads (8 waves), tile 128x128, K-step 32, double-buffered
// async-to-LDS tile staging (copy of tile t+1 overlaps compute of tile t).
// Wave grid 4(M) x 2(N): each wave owns a 32x64 sub-tile = 2x4 WMMA tiles.
// ---------------------------------------------------------------------------
template <bool F32OUT>
__global__ __launch_bounds__(256) void gemm_xwt(
    const _Float16* __restrict__ A,   // [M,K]
    const _Float16* __restrict__ W,   // [N,K]
    const float*    __restrict__ bias,// [N]
    _Float16*       __restrict__ Ch,  // [M,N] when !F32OUT
    float*          __restrict__ Cf,  // [M,N] when F32OUT
    int M, int N, int K) {
  constexpr int LDT = 40;                       // 32 + 8 pad, keeps 16B rows
  __shared__ __align__(16) _Float16 As[2][128 * LDT];
  __shared__ __align__(16) _Float16 Bs[2][128 * LDT];

  const int tid = threadIdx.x;
  const int wid = tid >> 5;
  const int m0  = blockIdx.y * 128;
  const int n0  = blockIdx.x * 128;
  const int wM  = (wid & 3) * 32;
  const int wN  = (wid >> 2) * 64;

  // per-thread slots of the cooperative tile copy (2 x 16B per array)
  const int row0 = tid >> 2;             // idx = tid
  const int seg0 = (tid & 3) * 8;
  const int row1 = (tid + 256) >> 2;     // idx = tid + 256
  const int seg1 = ((tid + 256) & 3) * 8;

  auto issue_tile = [&](int kc, int buf) {
    async_copy_b128(A + (size_t)(m0 + row0) * K + kc + seg0, &As[buf][row0 * LDT + seg0]);
    async_copy_b128(W + (size_t)(n0 + row0) * K + kc + seg0, &Bs[buf][row0 * LDT + seg0]);
    async_copy_b128(A + (size_t)(m0 + row1) * K + kc + seg1, &As[buf][row1 * LDT + seg1]);
    async_copy_b128(W + (size_t)(n0 + row1) * K + kc + seg1, &Bs[buf][row1 * LDT + seg1]);
  };

  const v8f vzero = {};
  v8f acc[2][4];
#pragma unroll
  for (int mi = 0; mi < 2; ++mi)
#pragma unroll
    for (int ni = 0; ni < 4; ++ni) acc[mi][ni] = vzero;

  const int nk = K >> 5;                 // K / 32 tiles
  issue_tile(0, 0);                      // prologue: tile 0 in flight

#pragma unroll 1
  for (int t = 0; t < nk; ++t) {
    const int buf = t & 1;
    if (t + 1 < nk) {
      issue_tile((t + 1) << 5, buf ^ 1); // stream next tile behind compute
      wait_async_le4();                  // in-order: tile t has landed
    } else {
      wait_async();
    }
    __syncthreads();

    v16h af[2], bf[4];
#pragma unroll
    for (int mi = 0; mi < 2; ++mi) af[mi] = frag_a_lds(As[buf], LDT, wM + mi * 16, 0);
#pragma unroll
    for (int ni = 0; ni < 4; ++ni) bf[ni] = frag_b_lds(Bs[buf], LDT, wN + ni * 16, 0);
#pragma unroll
    for (int mi = 0; mi < 2; ++mi)
#pragma unroll
      for (int ni = 0; ni < 4; ++ni)
        acc[mi][ni] = wmma_f16(af[mi], bf[ni], acc[mi][ni]);
    __syncthreads();
  }

  // C/D layout: element r -> row r (lanes 0-15) or 8+r (lanes 16-31), col=lane&15
  const int lane = tid & 31;
  const int cb   = lane & 15;
  const int rb   = (lane < 16) ? 0 : 8;
#pragma unroll
  for (int mi = 0; mi < 2; ++mi)
#pragma unroll
    for (int ni = 0; ni < 4; ++ni) {
      const int   col = n0 + wN + ni * 16 + cb;
      const float bv  = bias[col];
#pragma unroll
      for (int r = 0; r < 8; ++r) {
        const int   row = m0 + wM + mi * 16 + rb + r;
        const float v   = acc[mi][ni][r] + bv;
        if constexpr (F32OUT) Cf[(size_t)row * N + col] = v;
        else                  Ch[(size_t)row * N + col] = (_Float16)v;
      }
    }
}

// ---------------------------------------------------------------------------
// Fused causal GQA flash-attention.
// Grid: (T/64, H, B). Block: 128 threads = 4 waves; wave w owns 16 query rows.
// Per KV tile (64 keys): async K tile load overlapped with the coalesced
// V-transpose, S = Q K^T (16 WMMA), online softmax, O += P V (16 WMMA).
// Q2/K2/V2 layouts: [b*T, E] / [b*T, EKV] with head slice at h*128 / (h%8)*128.
// ---------------------------------------------------------------------------
__global__ __launch_bounds__(128) void attn_fused(
    const _Float16* __restrict__ Q2,
    const _Float16* __restrict__ K2,
    const _Float16* __restrict__ V2,
    _Float16*       __restrict__ Ah) {
  constexpr int LQ = 136;   // 128 + 8 pad
  constexpr int LV = 72;    // 64 + 8 pad
  constexpr int LP = 72;
  __shared__ __align__(16) _Float16 Qs[64 * LQ];
  __shared__ __align__(16) _Float16 Ks[64 * LQ];
  __shared__ __align__(16) _Float16 Vs[128 * LV];   // V transposed: [d][kv]
  __shared__ __align__(16) _Float16 Ps[4 * 16 * LP];// per-wave P staging

  const int tid  = threadIdx.x;
  const int lane = tid & 31;
  const int wid  = tid >> 5;
  const int qt   = blockIdx.x;            // query tile (64 rows)
  const int h    = blockIdx.y;
  const int b    = blockIdx.z;
  const int kh   = h & (HKQ - 1);         // GQA: kv head = h % 8
  const int q0   = qt * 64;

  const _Float16* Qg = Q2 + ((size_t)b * TQ + q0) * EQ + h * DQ;
  const _Float16* Kb = K2 + ((size_t)b * TQ) * EKVQ + kh * DQ;
  const _Float16* Vb = V2 + ((size_t)b * TQ) * EKVQ + kh * DQ;

  // stage the 64x128 Q tile (async), then hold Q fragments in registers
#pragma unroll
  for (int i = 0; i < 8; ++i) {
    const int idx = tid + i * 128;
    const int row = idx >> 4;
    const int seg = (idx & 15) * 8;
    async_copy_b128(Qg + (size_t)row * EQ + seg, Qs + row * LQ + seg);
  }
  wait_async();
  __syncthreads();

  v16h qf[4];
#pragma unroll
  for (int kt = 0; kt < 4; ++kt) qf[kt] = frag_a_lds(Qs, LQ, wid * 16, kt * 32);

  const v8f vzero = {};
  v8f  accO[8];
  float mrow[8], lrow[8];
#pragma unroll
  for (int dt = 0; dt < 8; ++dt) accO[dt] = vzero;
#pragma unroll
  for (int r = 0; r < 8; ++r) { mrow[r] = -1e30f; lrow[r] = 0.0f; }

  const float scale = 0.1767766952966369f;  // 1/sqrt(nheads=32), per reference
  const int   rb    = (lane < 16) ? 0 : 8;
  const int   cb    = lane & 15;
  _Float16*   ps    = Ps + wid * 16 * LP;

#pragma unroll 1
  for (int j = 0; j <= qt; ++j) {
    __syncthreads();   // previous iteration done reading Ks/Vs
    // K tile 64x128 row-major, async into LDS
#pragma unroll
    for (int i = 0; i < 8; ++i) {
      const int idx = tid + i * 128;
      const int row = idx >> 4;
      const int seg = (idx & 15) * 8;
      async_copy_b128(Kb + (size_t)(j * 64 + row) * EKVQ + seg,
                      Ks + row * LQ + seg);
    }
    // V tile transposed into Vs[d][kv]: coalesced b128 row-pair loads,
    // packed b32 column stores — overlaps with the async K copies above.
#pragma unroll
    for (int i = 0; i < 4; ++i) {
      const int idx = tid + i * 128;          // 512 tasks: 16 d-segs x 32 kv-pairs
      const int seg = (idx & 15) * 8;         // d segment base
      const int kvp = (idx >> 4) * 2;         // even kv
      union { uint4 q; _Float16 e[8]; } lo, hi;
      lo.q = *(const uint4*)(Vb + (size_t)(j * 64 + kvp) * EKVQ + seg);
      hi.q = *(const uint4*)(Vb + (size_t)(j * 64 + kvp + 1) * EKVQ + seg);
#pragma unroll
      for (int e = 0; e < 8; ++e) {
        union { _Float16 hh[2]; unsigned u; } pr;
        pr.hh[0] = lo.e[e];
        pr.hh[1] = hi.e[e];
        *(unsigned*)(Vs + (seg + e) * LV + kvp) = pr.u;   // ds_store_b32
      }
    }
    wait_async();
    __syncthreads();

    // ---- S = Q K^T : 4 kv sub-tiles x 4 K-steps -------------------------
    v8f s[4];
#pragma unroll
    for (int n = 0; n < 4; ++n) {
      s[n] = vzero;
#pragma unroll
      for (int kt = 0; kt < 4; ++kt)
        s[n] = wmma_f16(qf[kt], frag_b_lds(Ks, LQ, n * 16, kt * 32), s[n]);
    }
#pragma unroll
    for (int n = 0; n < 4; ++n)
#pragma unroll
      for (int r = 0; r < 8; ++r) s[n][r] *= scale;

    if (j == qt) {                          // causal mask, diagonal tile only
#pragma unroll
      for (int n = 0; n < 4; ++n) {
        const int col = j * 64 + n * 16 + cb;
#pragma unroll
        for (int r = 0; r < 8; ++r) {
          const int row = q0 + wid * 16 + rb + r;
          if (col > row) s[n][r] = -3.0e38f;
        }
      }
    }

    // ---- online softmax: rows live in lane-halves, reduce over 16 lanes --
#pragma unroll
    for (int r = 0; r < 8; ++r) {
      float mx = fmaxf(fmaxf(s[0][r], s[1][r]), fmaxf(s[2][r], s[3][r]));
#pragma unroll
      for (int msk = 1; msk < 16; msk <<= 1) mx = fmaxf(mx, __shfl_xor(mx, msk, 32));
      const float mnew = fmaxf(mrow[r], mx);
      const float corr = __expf(mrow[r] - mnew);
      mrow[r] = mnew;
      float rs = 0.0f;
#pragma unroll
      for (int n = 0; n < 4; ++n) {
        const float p = __expf(s[n][r] - mnew);
        s[n][r] = p;
        rs += p;
      }
#pragma unroll
      for (int msk = 1; msk < 16; msk <<= 1) rs += __shfl_xor(rs, msk, 32);
      lrow[r] = lrow[r] * corr + rs;
#pragma unroll
      for (int dt = 0; dt < 8; ++dt) accO[dt][r] *= corr;
    }

    // ---- re-layout P (C-layout -> A-fragment) through per-wave LDS -------
#pragma unroll
    for (int n = 0; n < 4; ++n)
#pragma unroll
      for (int r = 0; r < 8; ++r)
        ps[(rb + r) * LP + n * 16 + cb] = (_Float16)s[n][r];
    __syncthreads();

    // ---- O += P V : 8 d sub-tiles x 2 K-steps ---------------------------
    v16h pf[2];
#pragma unroll
    for (int kt = 0; kt < 2; ++kt) pf[kt] = frag_a_lds(ps, LP, 0, kt * 32);
#pragma unroll
    for (int dt = 0; dt < 8; ++dt)
#pragma unroll
      for (int kt = 0; kt < 2; ++kt)
        accO[dt] = wmma_f16(pf[kt], frag_b_lds(Vs, LV, dt * 16, kt * 32), accO[dt]);
  }

  // normalize by row-sum and store f16 output [b,t,h*128+d]
  _Float16* Og = Ah + ((size_t)b * TQ + q0) * EQ + h * DQ;
#pragma unroll
  for (int dt = 0; dt < 8; ++dt)
#pragma unroll
    for (int r = 0; r < 8; ++r) {
      const int row = wid * 16 + rb + r;
      Og[(size_t)row * EQ + dt * 16 + cb] = (_Float16)(accO[dt][r] / lrow[r]);
    }
}

// ---------------------------------------------------------------------------
// Host launcher
// ---------------------------------------------------------------------------
extern "C" void kernel_launch(void* const* d_in, const int* in_sizes, int n_in,
                              void* d_out, int out_size, void* d_ws, size_t ws_size,
                              hipStream_t stream) {
  (void)in_sizes; (void)n_in; (void)out_size; (void)ws_size;
  const float* x    = (const float*)d_in[0];
  const float* Wq_w = (const float*)d_in[1];
  const float* Wq_b = (const float*)d_in[2];
  const float* Wk_w = (const float*)d_in[3];
  const float* Wk_b = (const float*)d_in[4];
  const float* Wv_w = (const float*)d_in[5];
  const float* Wv_b = (const float*)d_in[6];
  const float* Wo_w = (const float*)d_in[7];
  const float* Wo_b = (const float*)d_in[8];

  const size_t nX  = (size_t)MTQ * EQ;     // 16.78M halves
  const size_t nWq = (size_t)EQ * EQ;
  const size_t nWk = (size_t)EKVQ * EQ;
  const size_t nKV = (size_t)MTQ * EKVQ;

  // Workspace (f16), stream-ordered aliasing to stay lean (~134 MB):
  //   Woh aliases Wqh (Wo converted after the Q GEMM consumed Wq),
  //   Ah  aliases Xh  (x no longer needed once Q/K/V GEMMs ran).
  _Float16* p   = (_Float16*)d_ws;
  _Float16* Xh  = p; p += nX;
  _Float16* Wqh = p; p += nWq;   _Float16* Woh = Wqh;
  _Float16* Wkh = p; p += nWk;
  _Float16* Wvh = p; p += nWk;
  _Float16* Q2  = p; p += nX;
  _Float16* K2  = p; p += nKV;
  _Float16* V2  = p; p += nKV;
  _Float16* Ah  = Xh;

  const dim3 blk(256);
  cvt_f32_to_f16<<<dim3((unsigned)(nX  / 1024)), blk, 0, stream>>>(x,    Xh,  (int)nX);
  cvt_f32_to_f16<<<dim3((unsigned)(nWq / 1024)), blk, 0, stream>>>(Wq_w, Wqh, (int)nWq);
  cvt_f32_to_f16<<<dim3((unsigned)(nWk / 1024)), blk, 0, stream>>>(Wk_w, Wkh, (int)nWk);
  cvt_f32_to_f16<<<dim3((unsigned)(nWk / 1024)), blk, 0, stream>>>(Wv_w, Wvh, (int)nWk);

  // Q projection, then overwrite Wq slot with f16(Wo)
  gemm_xwt<false><<<dim3(EQ / 128, MTQ / 128), blk, 0, stream>>>(
      Xh, Wqh, Wq_b, Q2, nullptr, MTQ, EQ, EQ);
  cvt_f32_to_f16<<<dim3((unsigned)(nWq / 1024)), blk, 0, stream>>>(Wo_w, Woh, (int)nWq);

  gemm_xwt<false><<<dim3(EKVQ / 128, MTQ / 128), blk, 0, stream>>>(
      Xh, Wkh, Wk_b, K2, nullptr, MTQ, EKVQ, EQ);
  gemm_xwt<false><<<dim3(EKVQ / 128, MTQ / 128), blk, 0, stream>>>(
      Xh, Wvh, Wv_b, V2, nullptr, MTQ, EKVQ, EQ);

  attn_fused<<<dim3(TQ / 64, HQ, BQ), dim3(128), 0, stream>>>(Q2, K2, V2, Ah);

  gemm_xwt<true><<<dim3(EQ / 128, MTQ / 128), blk, 0, stream>>>(
      Ah, Woh, Wo_b, nullptr, (float*)d_out, MTQ, EQ, EQ);
}
